// StatDynamicConv2d_6940667150581
// MI455X (gfx1250) — compile-verified, matrix operand order checked by reading, not run on hardware
//
#include <hip/hip_runtime.h>

// ---------------- problem constants ----------------
constexpr int NE    = 4;      // experts
constexpr int NCIN  = 128;
constexpr int NCOUT = 128;
constexpr int NSTAT = 6;
constexpr int NHID  = 16;
constexpr int NB    = 32;     // batch
constexpr int NH    = 64;
constexpr int NWD   = 64;
constexpr int KTOT  = NCIN * 9;        // 1152, K order: (ky*3+kx)*128 + cin
constexpr int PER_E = NCOUT * KTOT;    // 147456

// ---------------- vector types for WMMA ----------------
typedef __attribute__((ext_vector_type(16))) __bf16 v16bf;
typedef __attribute__((ext_vector_type(8)))  __bf16 v8bf;
typedef __attribute__((ext_vector_type(8)))  float  v8f;

// ================= router: Linear(6,16)->ReLU->Linear(16,4)->Softmax ======
__global__ void router_kernel(const float* __restrict__ stats,
                              const float* __restrict__ W1,
                              const float* __restrict__ b1,
                              const float* __restrict__ W2,
                              const float* __restrict__ b2,
                              float* __restrict__ routing)
{
    int b = threadIdx.x;
    if (b >= NB) return;
    float h[NHID];
#pragma unroll
    for (int j = 0; j < NHID; ++j) {
        float acc = b1[j];
#pragma unroll
        for (int s = 0; s < NSTAT; ++s) acc += stats[b * NSTAT + s] * W1[j * NSTAT + s];
        h[j] = acc > 0.f ? acc : 0.f;
    }
    float lg[NE];
    float mx = -1e30f;
#pragma unroll
    for (int e = 0; e < NE; ++e) {
        float acc = b2[e];
#pragma unroll
        for (int j = 0; j < NHID; ++j) acc += h[j] * W2[e * NHID + j];
        lg[e] = acc;
        mx = fmaxf(mx, acc);
    }
    float sum = 0.f;
#pragma unroll
    for (int e = 0; e < NE; ++e) { lg[e] = __expf(lg[e] - mx); sum += lg[e]; }
    float inv = 1.f / sum;
#pragma unroll
    for (int e = 0; e < NE; ++e) routing[b * NE + e] = lg[e] * inv;
}

// ===== per-sample expert mix -> bf16 dyn_w, K reordered to (r)*128+cin ====
__global__ void dynw_kernel(const float* __restrict__ we,
                            const float* __restrict__ routing,
                            __bf16* __restrict__ dynw)
{
    const int total = NB * PER_E;
    int i = blockIdx.x * blockDim.x + threadIdx.x;
    int stride = gridDim.x * blockDim.x;
    for (; i < total; i += stride) {
        int b    = i / PER_E;
        int rem  = i - b * PER_E;
        int cout = rem / KTOT;
        int k    = rem - cout * KTOT;
        int r    = k >> 7;          // ky*3+kx
        int cin  = k & 127;
        int wi   = cout * KTOT + cin * 9 + r;   // source: [cout][cin][ky][kx]
        const float* rt = routing + b * NE;
        float v = rt[0] * we[wi] + rt[1] * we[PER_E + wi]
                + rt[2] * we[2 * PER_E + wi] + rt[3] * we[3 * PER_E + wi];
        dynw[i] = (__bf16)v;
    }
}

// ================= dynamic bias ===========================================
__global__ void dynb_kernel(const float* __restrict__ be,
                            const float* __restrict__ routing,
                            float* __restrict__ dynb)
{
    int i = blockIdx.x * blockDim.x + threadIdx.x;
    if (i >= NB * NCOUT) return;
    int b  = i >> 7;
    int co = i & (NCOUT - 1);
    const float* rt = routing + b * NE;
    float v = 0.f;
#pragma unroll
    for (int e = 0; e < NE; ++e) v += rt[e] * be[e * NCOUT + co];
    dynb[i] = v;
}

// ================= implicit-GEMM conv, V_WMMA_F32_16X16X32_BF16 ===========
// Block (256 thr, 8 waves): sample b, two image rows -> tile [128 x 128 px].
// K-loop over 1152 in steps of 32, software-pipelined with LDS double
// buffering: async A-loads (ASYNCcnt) + im2col B-stores for tile k+1 run
// underneath the 8 WMMAs of tile k; s_wait_asynccnt 2 keeps next tile's DMA
// in flight while guaranteeing the current tile has landed.
__global__ void __launch_bounds__(256)
conv_wmma_kernel(const float* __restrict__ xg,
                 const __bf16* __restrict__ dynw,
                 const float* __restrict__ dynb,
                 float* __restrict__ out)
{
    // padded LDS tiles (row = 40 bf16 = 80 B; 16-B aligned chunk offsets)
    __shared__ __attribute__((aligned(16))) __bf16 lA[2][NCOUT][40];
    __shared__ __attribute__((aligned(16))) __bf16 lB[2][128][40];

    const int tid  = threadIdx.x;
    const int lane = tid & 31;
    const int wv   = tid >> 5;         // wave 0..7
    const int m    = lane & 15;        // M/N index within 16-tile
    const int half = lane >> 4;        // selects K half per ISA layout
    const int mt0  = (wv & 3) << 1;    // first of 2 M-tiles
    const int nt0  = (wv >> 2) << 2;   // first of 4 N-tiles
    const int y0   = blockIdx.x << 1;  // two image rows y0, y0+1
    const int b    = blockIdx.y;       // sample

    v8f acc[2][4];
#pragma unroll
    for (int mi = 0; mi < 2; ++mi)
#pragma unroll
        for (int t = 0; t < 4; ++t)
#pragma unroll
            for (int i = 0; i < 8; ++i) acc[mi][t][i] = 0.f;

    const __bf16* aBase = dynw + (size_t)b * NCOUT * KTOT;
    const float*  xBase = xg + (size_t)b * NCIN * NH * NWD;

    // ---- stage tile k0 into buffer pb: async A, im2col B ----
    auto stage = [&](int k0, int pb) {
#pragma unroll
        for (int h = 0; h < 2; ++h) {
            int c    = tid + (h << 8);          // 0..511
            int co   = c >> 2;
            int koff = (c & 3) << 3;
            const __bf16* g = aBase + (size_t)co * KTOT + k0 + koff;
            unsigned           ldsd = (unsigned)(uintptr_t)&lA[pb][co][koff];
            unsigned long long ga   = (unsigned long long)(uintptr_t)g;
            asm volatile("global_load_async_to_lds_b128 %0, %1, off"
                         :: "v"(ldsd), "v"(ga) : "memory");
        }
        const int r  = k0 >> 7;                 // ky*3+kx, uniform per step
        const int ky = r / 3;
        const int kx = r - 3 * ky;
        const int cinBase = (k0 & 127);
#pragma unroll
        for (int h = 0; h < 2; ++h) {
            int c   = tid + (h << 8);           // 0..511
            int pix = c >> 2;                   // 0..127
            int kk  = (c & 3) << 3;
            int yy  = y0 + (pix >> 6) + ky - 1;
            int xx  = (pix & 63) + kx - 1;
            bool ok = ((unsigned)yy < (unsigned)NH) && ((unsigned)xx < (unsigned)NWD);
            const float* src = xBase + ((size_t)(cinBase + kk) * NH + yy) * NWD + xx;
            v8bf pack;
#pragma unroll
            for (int i = 0; i < 8; ++i) {
                float v = ok ? src[(size_t)i * NH * NWD] : 0.f;  // plane stride
                pack[i] = (__bf16)v;
            }
            *(v8bf*)&lB[pb][pix][kk] = pack;
        }
    };

    // ---- 8 WMMAs on buffer pb (ISA 16-bit A/B fragment layouts) ----
    auto compute = [&](int pb) {
        v16bf afr[2];
#pragma unroll
        for (int mi = 0; mi < 2; ++mi) {
            const __bf16* arow = &lA[pb][((mt0 + mi) << 4) + m][0];
            v8bf lo = *(const v8bf*)(arow + (half << 3));        // K =  h*8..+7
            v8bf hi = *(const v8bf*)(arow + 16 + (half << 3));   // K = 16+h*8..+7
#pragma unroll
            for (int i = 0; i < 8; ++i) { afr[mi][i] = lo[i]; afr[mi][i + 8] = hi[i]; }
        }
#pragma unroll
        for (int t = 0; t < 4; ++t) {
            const __bf16* brow = &lB[pb][((nt0 + t) << 4) + m][0];
            v8bf lo = *(const v8bf*)(brow + (half << 3));
            v8bf hi = *(const v8bf*)(brow + 16 + (half << 3));
            v16bf bfr;
#pragma unroll
            for (int i = 0; i < 8; ++i) { bfr[i] = lo[i]; bfr[i + 8] = hi[i]; }
#pragma unroll
            for (int mi = 0; mi < 2; ++mi)
                acc[mi][t] = __builtin_amdgcn_wmma_f32_16x16x32_bf16(
                    /*neg_a=*/false, afr[mi], /*neg_b=*/false, bfr,
                    /*c_mod=*/(short)0, acc[mi][t],
                    /*reuse_a=*/false, /*reuse_b=*/false);
        }
    };

    // ---- software pipeline: prologue + 18 double-phase iterations ----
    stage(0, 0);
    for (int k0 = 0; k0 < KTOT; k0 += 64) {
        // phase 0: compute tile k0 (buf0), stage tile k0+32 (buf1)
        if (k0 + 32 < KTOT) {
            stage(k0 + 32, 1);
            asm volatile("s_wait_asynccnt 0x2" ::: "memory"); // current tile landed
        } else {
            asm volatile("s_wait_asynccnt 0x0" ::: "memory");
        }
        __syncthreads();
        compute(0);
        __syncthreads();                                      // buf0 free for reuse

        // phase 1: compute tile k0+32 (buf1), stage tile k0+64 (buf0)
        if (k0 + 64 < KTOT) {
            stage(k0 + 64, 0);
            asm volatile("s_wait_asynccnt 0x2" ::: "memory");
        } else {
            asm volatile("s_wait_asynccnt 0x0" ::: "memory");
        }
        __syncthreads();
        compute(1);
        __syncthreads();                                      // buf1 free for reuse
    }

    // ---- epilogue: D layout -> NCHW, add dynamic bias ----
#pragma unroll
    for (int mi = 0; mi < 2; ++mi) {
        float biasv[8];
#pragma unroll
        for (int v = 0; v < 8; ++v)
            biasv[v] = dynb[b * NCOUT + ((mt0 + mi) << 4) + v + (half << 3)];
#pragma unroll
        for (int t = 0; t < 4; ++t) {
            int pix = ((nt0 + t) << 4) + m;                  // 0..127
            int px  = (y0 + (pix >> 6)) * NWD + (pix & 63);
#pragma unroll
            for (int v = 0; v < 8; ++v) {
                int co = ((mt0 + mi) << 4) + v + (half << 3);
                out[((size_t)b * NCOUT + co) * (NH * NWD) + px] = acc[mi][t][v] + biasv[v];
            }
        }
    }
}

// ================= launcher ===============================================
extern "C" void kernel_launch(void* const* d_in, const int* in_sizes, int n_in,
                              void* d_out, int out_size, void* d_ws, size_t ws_size,
                              hipStream_t stream)
{
    const float* x     = (const float*)d_in[0]; // [32,128,64,64]
    const float* stats = (const float*)d_in[1]; // [32,6]
    const float* W1    = (const float*)d_in[2]; // [16,6]
    const float* b1    = (const float*)d_in[3]; // [16]
    const float* W2    = (const float*)d_in[4]; // [4,16]
    const float* b2    = (const float*)d_in[5]; // [4]
    const float* we    = (const float*)d_in[6]; // [4,128,128,3,3]
    const float* be    = (const float*)d_in[7]; // [4,128]
    float* out = (float*)d_out;

    char* ws = (char*)d_ws;
    float*  routing = (float*)ws;              // 512 B
    float*  dynb    = (float*)(ws + 1024);     // 16 KB
    __bf16* dynw    = (__bf16*)(ws + 32768);   // 9.44 MB (bf16, reordered K)

    router_kernel<<<1, 32, 0, stream>>>(stats, W1, b1, W2, b2, routing);

    dynw_kernel<<<4096, 256, 0, stream>>>(we, routing, dynw);

    dynb_kernel<<<(NB * NCOUT + 255) / 256, 256, 0, stream>>>(be, routing, dynb);

    dim3 grid(NH / 2, NB);   // (pair of image rows, sample)
    conv_wmma_kernel<<<grid, 256, 0, stream>>>(x, dynw, dynb, out);
}
